// MultiHeadCrossAttention_75488345195287
// MI455X (gfx1250) — matrix operand order repaired and checked by reading, not compiled, not run on hardware
//
#include <hip/hip_runtime.h>
#include <hip/hip_bf16.h>

#define CDIM 512
#define NHEADS 4
#define HD 128
#define LL 100   // H*W

typedef __attribute__((ext_vector_type(16))) _Float16 v16h;
typedef __attribute__((ext_vector_type(8)))  _Float16 v8h;
typedef __attribute__((ext_vector_type(8)))  float    v8f;

// A-fragment (16x32 f16, ISA 7.12.2): lane row m = lane%16, group g = lane/16.
// K order used here (not the ISA table order): one contiguous 16-run [16g, 16g+16).
__device__ __forceinline__ v16h load_a_frag(const _Float16* __restrict__ row,
                                            int c0, int g) {
    v8h lo = *(const v8h*)(row + c0 + 16 * g);       // 16B aligned
    v8h hi = *(const v8h*)(row + c0 + 16 * g + 8);   // 16B aligned
    return __builtin_shufflevector(lo, hi, 0, 1, 2, 3, 4, 5, 6, 7,
                                           8, 9, 10, 11, 12, 13, 14, 15);
}

// ---------------------------------------------------------------------------
// Tiled transpose + f32->f16: X [B*NIMG][512][100] -> Xh [B*NIMG*100][512]
// ---------------------------------------------------------------------------
template <int NIMG>
__global__ __launch_bounds__(256)
void transpose_kernel(const float* __restrict__ X, _Float16* __restrict__ Xh)
{
    __shared__ float tile[32 * LL];
    const int img = blockIdx.x;             // b*NIMG + n
    const int c0  = blockIdx.y * 32;
    const int tid = threadIdx.x;
    const float* src = X + ((size_t)img * CDIM + c0) * LL;
    for (int idx = tid; idx < 32 * LL; idx += 256)
        tile[idx] = src[idx];               // coalesced along l
    __syncthreads();
    _Float16* dst = Xh + (size_t)img * LL * CDIM + c0;
    for (int idx = tid; idx < 32 * LL; idx += 256) {
        int l = idx >> 5, c = idx & 31;     // contiguous along c on store
        dst[(size_t)l * CDIM + c] = (_Float16)tile[c * LL + l];
    }
}

// elementwise f32 -> f16 (weights)
__global__ __launch_bounds__(256)
void cvt_kernel(const float* __restrict__ src, _Float16* __restrict__ dst, int n)
{
    int i = blockIdx.x * 256 + threadIdx.x;
    if (i < n) dst[i] = (_Float16)src[i];
}

// ---------------------------------------------------------------------------
// QKV projection: Xh [4*TPER][512] f16, Wh [512][512] f16 (row-major [o][c]).
//   Y[t,o] = sum_c Xh[t,c]*Wh[o,c] + bias[o]
// VT==false: Y f16 [b][h][tl][hd]      (row-major, for Q/K)
// VT==true : Y f16 [b][h][hd][TPAD]    (transposed, for V; TPAD = Tk pad 32)
// one wave per 16x16 output tile; dual accumulators hide WMMA RAW hazards
// ---------------------------------------------------------------------------
template <int TPER, bool VT, int TPAD>
__global__ __launch_bounds__(32)
void proj_qkv_kernel(const _Float16* __restrict__ Xh,
                     const _Float16* __restrict__ Wh,
                     const float* __restrict__ bias,
                     _Float16* __restrict__ Y)
{
    const int Ttot = 4 * TPER;
    const int lane = threadIdx.x & 31;
    const int g    = lane >> 4;
    const int n16  = lane & 15;
    const int t0   = blockIdx.x * 16;
    const int o0   = blockIdx.y * 16;

    int tA = t0 + n16; if (tA >= Ttot) tA = Ttot - 1;
    const _Float16* xrow  = Xh + (size_t)tA * CDIM;
    const _Float16* wbase = Wh + (size_t)(o0 + n16) * CDIM;

    v8f acc0 = {}, acc1 = {};
    #pragma unroll 4
    for (int c0 = 0; c0 < CDIM; c0 += 64) {
        v16h a0 = load_a_frag(xrow, c0, g);
        v16h b0 = *(const v16h*)(wbase + c0 + 16 * g);        // 32B aligned
        v16h a1 = load_a_frag(xrow, c0 + 32, g);
        v16h b1 = *(const v16h*)(wbase + c0 + 32 + 16 * g);
        acc0 = __builtin_amdgcn_wmma_f32_16x16x32_f16(false, a0, false, b0,
                                                      (short)0, acc0, false, false);
        acc1 = __builtin_amdgcn_wmma_f32_16x16x32_f16(false, a1, false, b1,
                                                      (short)0, acc1, false, false);
    }
    v8f acc = acc0 + acc1;

    const int o  = o0 + n16;
    const int h  = o >> 7;
    const int hd = o & 127;
    const float bv = bias[o];
    #pragma unroll
    for (int r = 0; r < 8; ++r) {
        int t = t0 + r + 8 * g;              // C/D row = r + 8*g
        if (t < Ttot) {
            int b  = t / TPER;               // compile-time divisor
            int tl = t - b * TPER;
            size_t idx = VT
                ? ((size_t)(b * NHEADS + h) * HD + hd) * TPAD + tl
                : ((size_t)(b * NHEADS + h) * TPER + tl) * HD + hd;
            Y[idx] = (_Float16)(acc[r] + bv);
        }
    }
}

// ---------------------------------------------------------------------------
// Flash cross-attention for one (b,h), 16 query rows per wave.
// Q,K: f16 [bh][T][128] row-major; VT: f16 [bh][128][Tkpad] (transposed).
// outh: f16 [b][tq][512] written at column offset h*128 (merged heads).
// ---------------------------------------------------------------------------
__global__ __launch_bounds__(32)
void attn_kernel(const _Float16* __restrict__ Q, const _Float16* __restrict__ K,
                 const _Float16* __restrict__ VT, _Float16* __restrict__ outh,
                 int Tq, int Tk, int Tkpad, float scale)
{
    __shared__ __align__(64) _Float16 Pt[16 * 32]; // C-layout -> A-layout bridge

    const int lane = threadIdx.x & 31;
    const int g    = lane >> 4;
    const int n16  = lane & 15;
    const int bh   = blockIdx.y;
    const int b    = bh >> 2;
    const int h    = bh & 3;
    const int tq0  = blockIdx.x * 16;

    const _Float16* Qb = Q  + (size_t)bh * Tq * HD;
    const _Float16* Kb = K  + (size_t)bh * Tk * HD;
    const _Float16* Vb = VT + (size_t)bh * HD * Tkpad;

    // 16x128 Q row-block as 4 resident A-fragments
    int tA = tq0 + n16; if (tA >= Tq) tA = Tq - 1;
    const _Float16* qrow = Qb + (size_t)tA * HD;
    v16h qa[4];
    #pragma unroll
    for (int j = 0; j < 4; ++j) qa[j] = load_a_frag(qrow, 32 * j, g);

    // ---- pass 1: online row max / sum ----
    float mrun[8], lrun[8];
    #pragma unroll
    for (int r = 0; r < 8; ++r) { mrun[r] = -1e30f; lrun[r] = 0.f; }

    for (int k0 = 0; k0 < Tk; k0 += 16) {
        int kr = k0 + n16; bool kvalid = kr < Tk; if (!kvalid) kr = Tk - 1;
        const _Float16* krow = Kb + (size_t)kr * HD;
        v16h kb0 = *(const v16h*)(krow + 16 * g);              // 32B aligned
        v16h kb1 = *(const v16h*)(krow + 32 + 16 * g);
        v16h kb2 = *(const v16h*)(krow + 64 + 16 * g);
        v16h kb3 = *(const v16h*)(krow + 96 + 16 * g);
        v8f s0 = {}, s1 = {};
        s0 = __builtin_amdgcn_wmma_f32_16x16x32_f16(false, qa[0], false, kb0,
                                                    (short)0, s0, false, false);
        s1 = __builtin_amdgcn_wmma_f32_16x16x32_f16(false, qa[1], false, kb1,
                                                    (short)0, s1, false, false);
        s0 = __builtin_amdgcn_wmma_f32_16x16x32_f16(false, qa[2], false, kb2,
                                                    (short)0, s0, false, false);
        s1 = __builtin_amdgcn_wmma_f32_16x16x32_f16(false, qa[3], false, kb3,
                                                    (short)0, s1, false, false);
        v8f s = s0 + s1;
        #pragma unroll
        for (int r = 0; r < 8; ++r) {
            float sv = kvalid ? s[r] * scale : -1e30f;
            float tmax = sv;
            #pragma unroll
            for (int off = 1; off < 16; off <<= 1)
                tmax = fmaxf(tmax, __shfl_xor(tmax, off, 32));
            float mnew = fmaxf(mrun[r], tmax);
            float ex   = kvalid ? __expf(sv - mnew) : 0.f;
            float tsum = ex;
            #pragma unroll
            for (int off = 1; off < 16; off <<= 1)
                tsum += __shfl_xor(tsum, off, 32);
            lrun[r] = lrun[r] * __expf(mrun[r] - mnew) + tsum;
            mrun[r] = mnew;
        }
    }
    float linv[8];
    #pragma unroll
    for (int r = 0; r < 8; ++r) linv[r] = (lrun[r] > 0.f) ? 1.f / lrun[r] : 0.f;

    // ---- pass 2: recompute S, P = exp(s-m)/l, out += P @ V ----
    v8f oc[8];
    v8f zf = {};
    #pragma unroll
    for (int jn = 0; jn < 8; ++jn) oc[jn] = zf;

    for (int k0 = 0; k0 < Tk; k0 += 32) {
        #pragma unroll
        for (int sub = 0; sub < 2; ++sub) {
            int kr = k0 + sub * 16 + n16;
            bool kvalid = kr < Tk; if (!kvalid) kr = Tk - 1;
            const _Float16* krow = Kb + (size_t)kr * HD;
            v16h kb0 = *(const v16h*)(krow + 16 * g);
            v16h kb1 = *(const v16h*)(krow + 32 + 16 * g);
            v16h kb2 = *(const v16h*)(krow + 64 + 16 * g);
            v16h kb3 = *(const v16h*)(krow + 96 + 16 * g);
            v8f s0 = {}, s1 = {};
            s0 = __builtin_amdgcn_wmma_f32_16x16x32_f16(false, qa[0], false, kb0,
                                                        (short)0, s0, false, false);
            s1 = __builtin_amdgcn_wmma_f32_16x16x32_f16(false, qa[1], false, kb1,
                                                        (short)0, s1, false, false);
            s0 = __builtin_amdgcn_wmma_f32_16x16x32_f16(false, qa[2], false, kb2,
                                                        (short)0, s0, false, false);
            s1 = __builtin_amdgcn_wmma_f32_16x16x32_f16(false, qa[3], false, kb3,
                                                        (short)0, s1, false, false);
            v8f s = s0 + s1;
            #pragma unroll
            for (int r = 0; r < 8; ++r) {
                int m = r + 8 * g;
                float p = kvalid ? __expf(s[r] * scale - mrun[r]) * linv[r] : 0.f;
                Pt[m * 32 + sub * 16 + n16] = (_Float16)p;
            }
        }
        __syncthreads();
        v16h pa = load_a_frag(&Pt[n16 * 32], 0, g);   // ds_load_b128 x2
        #pragma unroll
        for (int jn = 0; jn < 8; ++jn) {
            // V^T: contiguous along tokens -> single aligned v16h per fragment
            v16h vb = *(const v16h*)(Vb + (size_t)(jn * 16 + n16) * Tkpad
                                        + k0 + 16 * g);
            oc[jn] = __builtin_amdgcn_wmma_f32_16x16x32_f16(false, pa, false, vb,
                                                            (short)0, oc[jn],
                                                            false, false);
        }
        __syncthreads();
    }

    #pragma unroll
    for (int jn = 0; jn < 8; ++jn) {
        int col = h * HD + jn * 16 + n16;
        #pragma unroll
        for (int r = 0; r < 8; ++r) {
            int t = tq0 + r + 8 * g;
            if (t < Tq)
                outh[((size_t)(b * Tq + t)) * CDIM + col] = (_Float16)oc[jn][r];
        }
    }
}

// ---------------------------------------------------------------------------
// Output projection: Xh f16 [4*TPER][512] (attention out, heads merged)
//   Out[b][n][o][l] = sum_c Xh[t,c]*Wh[o,c] + bias[o],  t = b*TPER + n*100 + l
// ---------------------------------------------------------------------------
template <int NIMG>
__global__ __launch_bounds__(32)
void proj_out_kernel(const _Float16* __restrict__ Xh,
                     const _Float16* __restrict__ Wh,
                     const float* __restrict__ bias,
                     float* __restrict__ Out)
{
    constexpr int TPER = NIMG * LL;
    const int Ttot = 4 * TPER;
    const int lane = threadIdx.x & 31;
    const int g    = lane >> 4;
    const int n16  = lane & 15;
    const int t0   = blockIdx.x * 16;
    const int o0   = blockIdx.y * 16;

    int tA = t0 + n16; if (tA >= Ttot) tA = Ttot - 1;
    const _Float16* xrow  = Xh + (size_t)tA * CDIM;
    const _Float16* wbase = Wh + (size_t)(o0 + n16) * CDIM;

    v8f acc0 = {}, acc1 = {};
    #pragma unroll 4
    for (int c0 = 0; c0 < CDIM; c0 += 64) {
        v16h a0 = load_a_frag(xrow, c0, g);
        v16h b0 = *(const v16h*)(wbase + c0 + 16 * g);
        v16h a1 = load_a_frag(xrow, c0 + 32, g);
        v16h b1 = *(const v16h*)(wbase + c0 + 32 + 16 * g);
        acc0 = __builtin_amdgcn_wmma_f32_16x16x32_f16(false, a0, false, b0,
                                                      (short)0, acc0, false, false);
        acc1 = __builtin_amdgcn_wmma_f32_16x16x32_f16(false, a1, false, b1,
                                                      (short)0, acc1, false, false);
    }
    v8f acc = acc0 + acc1;

    const int o = o0 + n16;
    const float bv = bias[o];
    #pragma unroll
    for (int r = 0; r < 8; ++r) {
        int t = t0 + r + 8 * g;
        if (t < Ttot) {
            int b  = t / TPER;               // compile-time divisors
            int tl = t - b * TPER;
            int nn = tl / LL;
            int l  = tl - nn * LL;
            Out[((size_t)(b * NIMG + nn) * CDIM + o) * LL + l] = acc[r] + bv;
        }
    }
}

// ---------------------------------------------------------------------------
extern "C" void kernel_launch(void* const* d_in, const int* in_sizes, int n_in,
                              void* d_out, int out_size, void* d_ws, size_t ws_size,
                              hipStream_t stream) {
    (void)in_sizes; (void)n_in; (void)out_size; (void)ws_size;
    const float* f1  = (const float*)d_in[0];
    const float* f2  = (const float*)d_in[1];
    const float* Wk1 = (const float*)d_in[2];  const float* bk1 = (const float*)d_in[3];
    const float* Wv1 = (const float*)d_in[4];  const float* bv1 = (const float*)d_in[5];
    const float* Wq2 = (const float*)d_in[6];  const float* bq2 = (const float*)d_in[7];
    // d_in[8]/d_in[9] (Wk2/bk2): K2 unused downstream in reference -> skipped
    const float* Wv2 = (const float*)d_in[10]; const float* bv2 = (const float*)d_in[11];
    const float* Wo1 = (const float*)d_in[12]; const float* bo1 = (const float*)d_in[13];
    const float* Wo2 = (const float*)d_in[14]; const float* bo2 = (const float*)d_in[15];

    char* ws = (char*)d_ws;                         // all offsets 256B-aligned
    _Float16* K1h  = (_Float16*)(ws + 0);           //  2,048,000 [16][ 500][128]
    _Float16* Q2h  = (_Float16*)(ws + 2048000);     // 30,720,000 [16][7500][128]
    _Float16* V1T  = (_Float16*)(ws + 32768000);    //  2,097,152 [16][128][ 512]
    _Float16* V2T  = (_Float16*)(ws + 34865152);    // 30,801,920 [16][128][7520]
    _Float16* X1h  = (_Float16*)(ws + 65667072);    //  2,048,000 [ 2000][512]
    _Float16* X2h  = (_Float16*)(ws + 67715072);    // 30,720,000 [30000][512]
    _Float16* Wk1h = (_Float16*)(ws + 98435072);    //  6 x 524,288
    _Float16* Wv1h = Wk1h + CDIM * CDIM;
    _Float16* Wq2h = Wv1h + CDIM * CDIM;
    _Float16* Wv2h = Wq2h + CDIM * CDIM;
    _Float16* Wo1h = Wv2h + CDIM * CDIM;
    _Float16* Wo2h = Wo1h + CDIM * CDIM;
    _Float16* o1h  = (_Float16*)(ws + 101580800);   //  2,048,000 [ 2000][512]
    _Float16* o2h  = (_Float16*)(ws + 103628800);   // 30,720,000 [30000][512]

    const dim3 b32(32, 1, 1);
    const dim3 b256(256, 1, 1);
    const float scale = 0.08838834764831845f;       // 128^-0.5
    const int   WN = CDIM * CDIM;

    // stage 0: layout transforms (f32 -> f16, token-major / weight convert)
    transpose_kernel<5> <<<dim3(20, 16, 1),  b256, 0, stream>>>(f1, X1h);
    transpose_kernel<75><<<dim3(300, 16, 1), b256, 0, stream>>>(f2, X2h);
    cvt_kernel<<<dim3(WN / 256, 1, 1), b256, 0, stream>>>(Wk1, Wk1h, WN);
    cvt_kernel<<<dim3(WN / 256, 1, 1), b256, 0, stream>>>(Wv1, Wv1h, WN);
    cvt_kernel<<<dim3(WN / 256, 1, 1), b256, 0, stream>>>(Wq2, Wq2h, WN);
    cvt_kernel<<<dim3(WN / 256, 1, 1), b256, 0, stream>>>(Wv2, Wv2h, WN);
    cvt_kernel<<<dim3(WN / 256, 1, 1), b256, 0, stream>>>(Wo1, Wo1h, WN);
    cvt_kernel<<<dim3(WN / 256, 1, 1), b256, 0, stream>>>(Wo2, Wo2h, WN);

    // stage 1: projections (K/Q row-major, V transposed)
    proj_qkv_kernel< 500, false,  512><<<dim3(125, 32, 1),  b32, 0, stream>>>(X1h, Wk1h, bk1, K1h);
    proj_qkv_kernel< 500, true,   512><<<dim3(125, 32, 1),  b32, 0, stream>>>(X1h, Wv1h, bv1, V1T);
    proj_qkv_kernel<7500, false, 7520><<<dim3(1875, 32, 1), b32, 0, stream>>>(X2h, Wq2h, bq2, Q2h);
    proj_qkv_kernel<7500, true,  7520><<<dim3(1875, 32, 1), b32, 0, stream>>>(X2h, Wv2h, bv2, V2T);

    // stage 2: cross attention (two directions over the shared logit structure)
    attn_kernel<<<dim3(469, 16, 1), b32, 0, stream>>>(Q2h, K1h, V1T, o2h, 7500, 500, 512, scale);
    attn_kernel<<<dim3(32, 16, 1),  b32, 0, stream>>>(K1h, Q2h, V2T, o1h, 500, 7500, 7520, scale);

    // stage 3: output projections, scatter into [B,n,C,H,W]; out1 then out2
    float* out1 = (float*)d_out;
    float* out2 = out1 + (size_t)4 * 5 * CDIM * LL;
    proj_out_kernel<5> <<<dim3(125, 32, 1),  b32, 0, stream>>>(o1h, Wo1h, bo1, out1);
    proj_out_kernel<75><<<dim3(1875, 32, 1), b32, 0, stream>>>(o2h, Wo2h, bo2, out2);
}
